// MSARowAttentionWithPairBias_67912022885140
// MI455X (gfx1250) — matrix operand order repaired and checked by reading, not compiled, not run on hardware
//
#include <hip/hip_runtime.h>

#define S_DIM 128
#define I_DIM 256
#define J_DIM 256
#define C_DIM 32
#define H_DIM 8
#define HC    256

typedef __attribute__((ext_vector_type(16))) __bf16 v16bf;
typedef __attribute__((ext_vector_type(8)))  float  v8f;
typedef __attribute__((ext_vector_type(4)))  float  v4f;
typedef __attribute__((ext_vector_type(4)))  unsigned int v4u;
typedef __attribute__((ext_vector_type(4)))  int v4i;

typedef __attribute__((address_space(1))) v4i gas_v4i;   // global int4
typedef __attribute__((address_space(3))) v4i las_v4i;   // LDS int4

union BF16Frag { v16bf v; v4u q[2]; };
union BF16x8   { v4u q; __bf16 e[8]; };
union F32x8    { v8f v; v4f q[2]; };

#if __has_builtin(__builtin_amdgcn_global_load_async_to_lds_b128)
#define HAVE_ASYNC_LDS 1
#else
#define HAVE_ASYNC_LDS 0
#endif

__device__ __forceinline__ float fast_rcp(float x) {
#if __has_builtin(__builtin_amdgcn_rcpf)
  return __builtin_amdgcn_rcpf(x);
#else
  return 1.0f / x;
#endif
}
__device__ __forceinline__ float fast_rsqrt(float x) {
#if __has_builtin(__builtin_amdgcn_rsqf)
  return __builtin_amdgcn_rsqf(x);
#else
  return rsqrtf(x);
#endif
}

__device__ __forceinline__ v8f wmma_bf16f32(v16bf a, v16bf b, v8f c) {
  return __builtin_amdgcn_wmma_f32_16x16x32_bf16(false, a, false, b, (short)0, c,
                                                 false, false);
}

__device__ __forceinline__ v16bf load_frag_2x16B(const __bf16* p0, const __bf16* p1) {
  BF16Frag f;
  f.q[0] = *(const v4u*)p0;
  f.q[1] = *(const v4u*)p1;
  return f.v;
}

__device__ __forceinline__ float wave_sum32(float x) {
#pragma unroll
  for (int m = 16; m >= 1; m >>= 1) x += __shfl_xor(x, m, 32);
  return x;
}

// ---------------------------------------------------------------------------
// Kernel 1: convert weights to bf16, transposed to [n][k] for B-fragment loads
// ---------------------------------------------------------------------------
__global__ void __launch_bounds__(256) prep_weights(
    const float* __restrict__ wq, const float* __restrict__ wk,
    const float* __restrict__ wv, const float* __restrict__ wg,
    const float* __restrict__ wo,
    __bf16* __restrict__ wqT, __bf16* __restrict__ wkT,
    __bf16* __restrict__ wvT, __bf16* __restrict__ wgT,
    __bf16* __restrict__ woT) {
  int tid = blockIdx.x * 256 + threadIdx.x;
  if (tid < 4 * 8192) {
    int w = tid >> 13, r = tid & 8191;
    int n = r >> 5, kk = r & 31;                    // n<256 cols, kk<32 rows
    const float* src = (w == 0) ? wq : (w == 1) ? wk : (w == 2) ? wv : wg;
    __bf16* dst = (w == 0) ? wqT : (w == 1) ? wkT : (w == 2) ? wvT : wgT;
    dst[n * 32 + kk] = (__bf16)src[kk * 256 + n];
  } else if (tid < 5 * 8192) {
    int r = tid - 4 * 8192;
    int n = r & 31, kk = r >> 5;                    // wo is (256,32): woT[n][kk]
    woT[n * 256 + kk] = (__bf16)wo[kk * 32 + n];
  }
}

// ---------------------------------------------------------------------------
// Kernel 2: layernorm(m) + q/k/v/gate projections (WMMA bf16, K=32 one-shot)
// q (C^-0.5 folded), k, gate(sigmoid) -> [s][h][i][c] via per-wave LDS tile
// bounce so every lane emits one global_store_b128 per fragment.
// v -> [s][h][c][i] (i-contiguous per lane) via direct packed b128 stores.
// ---------------------------------------------------------------------------
__global__ void __launch_bounds__(256) ln_proj_kernel(
    const float* __restrict__ m, const float* __restrict__ ln_scale,
    const float* __restrict__ ln_bias,
    const __bf16* __restrict__ wqT, const __bf16* __restrict__ wkT,
    const __bf16* __restrict__ wvT, const __bf16* __restrict__ wgT,
    __bf16* __restrict__ qo, __bf16* __restrict__ ko,
    __bf16* __restrict__ vo, __bf16* __restrict__ go) {
  __shared__ __bf16 a_lds[16 * 32];
  __shared__ __bf16 st_lds[8 * 256];                // per-wave 16x16 store tile
  int tid = threadIdx.x, wave = tid >> 5, lane = tid & 31;
  int r0 = blockIdx.x * 16;                         // global row (s*I + i)
  float sc = ln_scale[lane], bi = ln_bias[lane];
#pragma unroll
  for (int rr = 0; rr < 2; ++rr) {                  // wave handles 2 rows
    int rloc = wave + rr * 8;
    float x = m[(size_t)(r0 + rloc) * 32 + lane];
    float mu = wave_sum32(x) * (1.0f / 32.0f);
    float d = x - mu;
    float var = wave_sum32(d * d) * (1.0f / 32.0f);
    float y = d * fast_rsqrt(var + 1e-5f) * sc + bi;
    a_lds[rloc * 32 + lane] = (__bf16)y;
  }
  __syncthreads();
  int lrow = lane & 15, lhi = lane >> 4;
  v16bf a = load_frag_2x16B(&a_lds[lrow * 32 + lhi * 8],
                            &a_lds[lrow * 32 + lhi * 8 + 16]);
  int s = r0 >> 8;
  int i = r0 & 255;
  __bf16* tw = st_lds + wave * 256;
  int row2 = lane >> 1, half = lane & 1;            // store-phase mapping
#pragma unroll
  for (int w = 0; w < 4; ++w) {                     // compile-time weight idx
    const __bf16* wT = (w == 0) ? wqT : (w == 1) ? wkT : (w == 2) ? wvT : wgT;
    __bf16* dst = (w == 0) ? qo : (w == 1) ? ko : (w == 2) ? vo : go;
#pragma unroll
    for (int tt = 0; tt < 2; ++tt) {
      int n0 = (wave * 2 + tt) * 16;                // 16 n-tiles / weight
      const __bf16* bp = wT + (n0 + lrow) * 32 + lhi * 8;
      BF16Frag bf; bf.q[0] = *(const v4u*)bp; bf.q[1] = *(const v4u*)(bp + 16);
      v8f c = {};
      c = wmma_bf16f32(a, bf.v, c);
      int h = n0 >> 5, cc0 = n0 & 31;               // tile-constant
      if (w == 2) {
        // V: [s][h][c][i]; lane column cc, rows i-contiguous -> pack b128
        int cc = cc0 + lrow;
        BF16x8 st;
#pragma unroll
        for (int r = 0; r < 8; ++r) st.e[r] = (__bf16)c[r];
        __bf16* dp = dst + ((size_t)(s * H_DIM + h) * C_DIM + cc) * I_DIM
                     + i + lhi * 8;
        *(v4u*)dp = st.q;
      } else {
        // q/k/gate: bounce C-layout tile through LDS, then b128 row stores
#pragma unroll
        for (int r = 0; r < 8; ++r) {
          float val = c[r];
          if (w == 0) val *= 0.17677669529663687f;  // C^-0.5 folded into q
          if (w == 3) val = fast_rcp(1.0f + __expf(-val));
          tw[(r + lhi * 8) * 16 + lrow] = (__bf16)val;
        }
        asm volatile("s_wait_dscnt 0" ::: "memory");
        v4u dq = *(const v4u*)(tw + row2 * 16 + half * 8);
        __bf16* dp = dst + ((size_t)(s * H_DIM + h) * I_DIM + i + row2) * 32
                     + cc0 + half * 8;
        *(v4u*)dp = dq;
      }
    }
  }
}

// ---------------------------------------------------------------------------
// Kernel 3: layernorm(z) @ wb -> bias[h][i][j] (f32). One wave per (i,j) row.
// ---------------------------------------------------------------------------
__global__ void __launch_bounds__(256) bias_kernel(
    const float* __restrict__ z, const float* __restrict__ lnb_scale,
    const float* __restrict__ lnb_bias, const float* __restrict__ wb,
    float* __restrict__ bias) {
  int tid = threadIdx.x, wave = tid >> 5, lane = tid & 31;
  int pair = blockIdx.x * 8 + wave;                 // i*256 + j
  float x = z[(size_t)pair * 32 + lane];
  float mu = wave_sum32(x) * (1.0f / 32.0f);
  float d = x - mu;
  float var = wave_sum32(d * d) * (1.0f / 32.0f);
  float y = d * fast_rsqrt(var + 1e-5f) * lnb_scale[lane] + lnb_bias[lane];
#pragma unroll
  for (int h = 0; h < 8; ++h) {
    float p = wave_sum32(y * wb[lane * 8 + h]);
    if (lane == 0) bias[(size_t)h * 65536 + pair] = p;
  }
}

// ---------------------------------------------------------------------------
// Kernel 4: fused attention per (s,h), TRANSPOSED scores: S^T = K·Q^T.
// Each lane owns one i column -> scalar softmax state, 1 shfl per reduction.
// K, V^T staged via async global->LDS DMA. P^T bounce uses packed b128 DS ops.
// O^T gate/store fully 16B-vectorized.
// ---------------------------------------------------------------------------
__global__ void __launch_bounds__(256) attn_kernel(
    const __bf16* __restrict__ q, const __bf16* __restrict__ k,
    const __bf16* __restrict__ v, const __bf16* __restrict__ gate,
    const float* __restrict__ bias, __bf16* __restrict__ out_g) {
  __shared__ __bf16 k_lds[256 * 32];                // [j][c]
  __shared__ __bf16 vT_lds[32 * 256];               // [c][j] (as stored)
  __shared__ __bf16 p_lds[8 * 16 * 32];             // per-wave P^T [i][j32]
  int sh = blockIdx.x;
  int s = sh >> 3, h = sh & 7;
  int tid = threadIdx.x, wave = tid >> 5, lane = tid & 31;
  int lrow = lane & 15, lhi = lane >> 4;
  size_t base = (size_t)(s * H_DIM + h) * I_DIM * 32;
#if HAVE_ASYNC_LDS
  {
    const __bf16* ksrc = k + base;
    const __bf16* vsrc = v + base;                  // already [c][i]
#pragma unroll
    for (int it = 0; it < 4; ++it) {
      int idx = tid + it * 256;                     // 1024 chunks of 16B each
      __builtin_amdgcn_global_load_async_to_lds_b128(
          (gas_v4i*)(ksrc + idx * 8), (las_v4i*)(k_lds + idx * 8), 0, 0);
      __builtin_amdgcn_global_load_async_to_lds_b128(
          (gas_v4i*)(vsrc + idx * 8), (las_v4i*)(vT_lds + idx * 8), 0, 0);
    }
    asm volatile("s_wait_asynccnt 0" ::: "memory");
  }
#else
  {
    const v4u* ks = (const v4u*)(k + base);
    const v4u* vs = (const v4u*)(v + base);
    v4u* kd = (v4u*)k_lds;
    v4u* vd = (v4u*)vT_lds;
    for (int idx = tid; idx < 1024; idx += 256) { kd[idx] = ks[idx]; vd[idx] = vs[idx]; }
  }
#endif
  __syncthreads();

  __bf16* pw = p_lds + wave * 512;
#pragma unroll 1
  for (int ii = 0; ii < 2; ++ii) {
    int i0 = (wave * 2 + ii) * 16;
    // Q^T B-fragment straight from global [i][c]: lane col i=i0+lrow
    const __bf16* qp = q + base + (size_t)(i0 + lrow) * 32 + lhi * 8;
    v16bf qb = load_frag_2x16B(qp, qp + 16);
    float rm = -1e30f, rl = 0.0f;                   // scalar per lane (one i)
    v8f o0 = {}, o1 = {};                           // O^T tiles: c 0-15, 16-31
    const float* bb = bias + (size_t)h * 65536 + (size_t)(i0 + lrow) * 256;
#pragma unroll 1
    for (int j0 = 0; j0 < 256; j0 += 32) {
      if (j0 + 32 < 256)
        __builtin_prefetch(bb + j0 + 32, 0, 0);     // next bias chunk
      // bias as C operand: contiguous in j per lane -> vector loads
      F32x8 cb0, cb1;
      cb0.q[0] = *(const v4f*)(bb + j0 + lhi * 8);
      cb0.q[1] = *(const v4f*)(bb + j0 + lhi * 8 + 4);
      cb1.q[0] = *(const v4f*)(bb + j0 + 16 + lhi * 8);
      cb1.q[1] = *(const v4f*)(bb + j0 + 16 + lhi * 8 + 4);
      // K A-fragments: lane row j
      const __bf16* ka0 = k_lds + (j0 + lrow) * 32 + lhi * 8;
      const __bf16* ka1 = k_lds + (j0 + 16 + lrow) * 32 + lhi * 8;
      v16bf ak0 = load_frag_2x16B(ka0, ka0 + 16);
      v16bf ak1 = load_frag_2x16B(ka1, ka1 + 16);
      v8f s0 = wmma_bf16f32(ak0, qb, cb0.v);        // S^T: M=j, N=i
      v8f s1 = wmma_bf16f32(ak1, qb, cb1.v);
      // online softmax over j: in-register + one cross-half shuffle
      float cm = fmaxf(s0[0], s1[0]);
#pragma unroll
      for (int r = 1; r < 8; ++r) cm = fmaxf(cm, fmaxf(s0[r], s1[r]));
      cm = fmaxf(cm, __shfl_xor(cm, 16, 32));
      float mn2 = fmaxf(rm, cm);
      float corr = __expf(rm - mn2);
      rm = mn2;
      float p0[8], p1[8], cs = 0.0f;
#pragma unroll
      for (int r = 0; r < 8; ++r) {
        p0[r] = __expf(s0[r] - mn2);
        p1[r] = __expf(s1[r] - mn2);
        cs += p0[r] + p1[r];
      }
      cs += __shfl_xor(cs, 16, 32);
      rl = rl * corr + cs;
#pragma unroll
      for (int r = 0; r < 8; ++r) { o0[r] *= corr; o1[r] *= corr; }
      // P^T -> per-wave LDS [i_local][j32], packed b128 stores
      BF16x8 pk0, pk1;
#pragma unroll
      for (int r = 0; r < 8; ++r) { pk0.e[r] = (__bf16)p0[r]; pk1.e[r] = (__bf16)p1[r]; }
      *(v4u*)(pw + lrow * 32 + lhi * 8)      = pk0.q;
      *(v4u*)(pw + lrow * 32 + 16 + lhi * 8) = pk1.q;
      asm volatile("s_wait_dscnt 0" ::: "memory");
      // P^T B-fragment: lane col i=lrow, contiguous j
      const __bf16* pp = pw + lrow * 32 + lhi * 8;
      v16bf pb = load_frag_2x16B(pp, pp + 16);
      // V^T A-fragments: lane row c
      const __bf16* va0 = vT_lds + lrow * 256 + j0 + lhi * 8;
      const __bf16* va1 = vT_lds + (16 + lrow) * 256 + j0 + lhi * 8;
      v16bf av0 = load_frag_2x16B(va0, va0 + 16);
      v16bf av1 = load_frag_2x16B(va1, va1 + 16);
      o0 = wmma_bf16f32(av0, pb, o0);               // O^T: M=c, N=i
      o1 = wmma_bf16f32(av1, pb, o1);
    }
    // finalize: normalize, gate ([i][c] -> contiguous c per lane), pack, store
    float inv = fast_rcp(rl);
    const __bf16* gg = gate + base + (size_t)(i0 + lrow) * 32;
    BF16x8 g0, g1, ob0, ob1;
    g0.q = *(const v4u*)(gg + lhi * 8);
    g1.q = *(const v4u*)(gg + 16 + lhi * 8);
#pragma unroll
    for (int r = 0; r < 8; ++r) {
      ob0.e[r] = (__bf16)(o0[r] * inv * (float)g0.e[r]);
      ob1.e[r] = (__bf16)(o1[r] * inv * (float)g1.e[r]);
    }
    __bf16* og = out_g + ((size_t)(s * I_DIM) + i0 + lrow) * 256 + h * 32;
    *(v4u*)(og + lhi * 8)      = ob0.q;
    *(v4u*)(og + 16 + lhi * 8) = ob1.q;
  }
}

// ---------------------------------------------------------------------------
// Kernel 5: out = x(32768x256 bf16) @ wo(256x32) + bo -> f32
// ---------------------------------------------------------------------------
__global__ void __launch_bounds__(256) out_proj_kernel(
    const __bf16* __restrict__ x, const __bf16* __restrict__ woT,
    const float* __restrict__ bo, float* __restrict__ out) {
  int tid = threadIdx.x, wave = tid >> 5, lane = tid & 31;
  int lrow = lane & 15, lhi = lane >> 4;
  int i0 = blockIdx.x * 128 + wave * 16;
#pragma unroll 1
  for (int nt = 0; nt < 2; ++nt) {
    float bval = bo[nt * 16 + lrow];
    v8f c;
#pragma unroll
    for (int r = 0; r < 8; ++r) c[r] = bval;
#pragma unroll 1
    for (int k0 = 0; k0 < 256; k0 += 32) {
      const __bf16* ap = x + (size_t)(i0 + lrow) * 256 + k0 + lhi * 8;
      v16bf a = load_frag_2x16B(ap, ap + 16);
      const __bf16* bp = woT + (nt * 16 + lrow) * 256 + k0 + lhi * 8;
      BF16Frag f; f.q[0] = *(const v4u*)bp; f.q[1] = *(const v4u*)(bp + 16);
      c = wmma_bf16f32(a, f.v, c);
    }
    float* op = out + (size_t)(i0 + lhi * 8) * 32 + nt * 16 + lrow;
#pragma unroll
    for (int r = 0; r < 8; ++r) op[r * 32] = c[r];
  }
}

// ---------------------------------------------------------------------------
extern "C" void kernel_launch(void* const* d_in, const int* in_sizes, int n_in,
                              void* d_out, int out_size, void* d_ws, size_t ws_size,
                              hipStream_t stream) {
  const float* m         = (const float*)d_in[0];
  const float* z         = (const float*)d_in[1];
  const float* ln_scale  = (const float*)d_in[2];
  const float* ln_bias   = (const float*)d_in[3];
  const float* lnb_scale = (const float*)d_in[4];
  const float* lnb_bias  = (const float*)d_in[5];
  const float* wq        = (const float*)d_in[6];
  const float* wk        = (const float*)d_in[7];
  const float* wv        = (const float*)d_in[8];
  const float* wb        = (const float*)d_in[9];
  const float* wg        = (const float*)d_in[10];
  const float* wo        = (const float*)d_in[11];
  const float* bo        = (const float*)d_in[12];
  float* out = (float*)d_out;

  char* ws = (char*)d_ws;
  __bf16* wqT = (__bf16*)(ws + 0);
  __bf16* wkT = (__bf16*)(ws + 16384);
  __bf16* wvT = (__bf16*)(ws + 32768);
  __bf16* wgT = (__bf16*)(ws + 49152);
  __bf16* woT = (__bf16*)(ws + 65536);
  size_t off = 81920;
  __bf16* qb   = (__bf16*)(ws + off); off += 16777216;   // [s][h][i][c]
  __bf16* kb   = (__bf16*)(ws + off); off += 16777216;   // [s][h][i][c]
  __bf16* vb   = (__bf16*)(ws + off); off += 16777216;   // [s][h][c][i]
  __bf16* gb   = (__bf16*)(ws + off); off += 16777216;   // [s][h][i][c]
  float*  bias = (float*)(ws + off);  off += 2097152;    // [h][i][j]
  __bf16* xg   = (__bf16*)(ws + off); off += 16777216;   // [s*i][h*c]

  prep_weights<<<160, 256, 0, stream>>>(wq, wk, wv, wg, wo, wqT, wkT, wvT, wgT, woT);
  ln_proj_kernel<<<2048, 256, 0, stream>>>(m, ln_scale, ln_bias, wqT, wkT, wvT, wgT,
                                           qb, kb, vb, gb);
  bias_kernel<<<8192, 256, 0, stream>>>(z, lnb_scale, lnb_bias, wb, bias);
  attn_kernel<<<1024, 256, 0, stream>>>(qb, kb, vb, gb, bias, xg);
  out_proj_kernel<<<256, 256, 0, stream>>>(xg, woT, bo, out);
}
